// Attention_19524921327779
// MI455X (gfx1250) — matrix operand order repaired and checked by reading, not compiled, not run on hardware
//
#include <hip/hip_runtime.h>
#include <hip/hip_bf16.h>

// ---------------------------------------------------------------------------
// MHA forward for MI455X (gfx1250, wave32, WMMA 16x16x32 bf16).
//   B=4, N=2048, E=512, H=8, D=64.
// Pipeline:
//   0) elementwise f32->bf16 pre-pass for x, qkv_w, proj_w (hardware cvt)
//   1) qkv GEMM (double-buffered bf16 fragments + WMMA)
//        -> Q,K [B,H,N,D], V^T [B,H,D,N]
//   2) flash attention (streaming softmax; K/V L2-resident; V loads hoisted
//        ahead of the S-WMMAs so softmax VALU hides their latency)
//   3) output projection (double-buffered bf16 + WMMA), f32 out
// Scratch: Q,K,VT,O (4 x 8 MB) + Xbf (8 MB) + Wq (1.5 MB) + Wp (0.5 MB).
// ---------------------------------------------------------------------------

typedef __attribute__((ext_vector_type(16))) __bf16          v16bf;
typedef __attribute__((ext_vector_type(4)))  __bf16          v4bf;
typedef __attribute__((ext_vector_type(8)))  float           v8f;
typedef __attribute__((ext_vector_type(4)))  float           v4f;
typedef __attribute__((ext_vector_type(16))) unsigned short  v16u;
typedef __attribute__((ext_vector_type(8)))  unsigned short  v8u;

// A/B fragment from two 8-element contiguous bf16 runs (16B loads each).
__device__ __forceinline__ v16bf ld_bf16_2x8(const __bf16* p0, const __bf16* p1) {
  v8u a = *(const v8u*)p0;
  v8u b = *(const v8u*)p1;
  v16u u;
#pragma unroll
  for (int i = 0; i < 8; ++i) { u[i] = a[i]; u[8 + i] = b[i]; }
  return __builtin_bit_cast(v16bf, u);
}

__device__ __forceinline__ v16bf ld_bf16_16(const __bf16* p) {
  return ld_bf16_2x8(p, p + 8);
}

__device__ __forceinline__ v8f wmma_bf16(v16bf a, v16bf b, v8f c) {
  return __builtin_amdgcn_wmma_f32_16x16x32_bf16(false, a, false, b,
                                                 (short)0, c, false, false);
}

// ---------------------------------------------------------------------------
// Kernel 0: elementwise f32 -> bf16 (hardware v_cvt, 4 elements/thread).
// ---------------------------------------------------------------------------
__global__ __launch_bounds__(256) void cvt_bf16_kernel(
    const float* __restrict__ in, __bf16* __restrict__ out, int n) {
  const int i = (blockIdx.x * 256 + threadIdx.x) * 4;
  if (i < n) {
    v4f f = *(const v4f*)(in + i);
    v4bf o;
#pragma unroll
    for (int j = 0; j < 4; ++j) o[j] = (__bf16)f[j];
    *(v4bf*)(out + i) = o;
  }
}

// ---------------------------------------------------------------------------
// Kernel A: qkv = x @ qkv_w^T + qkv_b   (M=8192, Nc=1536, K=512)
// Each wave: 16 rows x 64 cols.  Block = 4 waves -> 16 x 256 tile.
// K-chunk fragments are double-buffered so the 4 WMMAs of chunk k overlap
// with the loads of chunk k+1 (partial loadcnt waits instead of ==0).
// Epilogue scatters into Q (pre-scaled by 1/sqrt(D)), K, and V^T (bf16).
// ---------------------------------------------------------------------------
__global__ __launch_bounds__(128) void qkv_kernel(
    const __bf16* __restrict__ Xb, const __bf16* __restrict__ Wq,
    const float* __restrict__ qkv_b, __bf16* __restrict__ Qs,
    __bf16* __restrict__ Ks, __bf16* __restrict__ VT) {
  const int lane = threadIdx.x & 31;
  const int wave = threadIdx.x >> 5;
  const int l16  = lane & 15;
  const int hf   = lane >> 4;                  // half-wave id
  const int b0   = hf * 16, b1 = hf * 16 + 8;  // A-frag K-run bases (contiguous k = 16h + s)
  const int rowbase = blockIdx.x * 16;
  const int colbase = blockIdx.y * 256 + wave * 64;

  v8f acc[4] = {{}, {}, {}, {}};
  const __bf16* xrow = Xb + (size_t)(rowbase + l16) * 512;

  auto loadA = [&](int kc) {
    return ld_bf16_2x8(xrow + kc + b0, xrow + kc + b1);
  };
  auto loadB = [&](int kc, int t) {
    return ld_bf16_16(Wq + (size_t)(colbase + t * 16 + l16) * 512 + kc +
                      hf * 16);
  };

  v16bf a0 = loadA(0);
  v16bf w0[4];
#pragma unroll
  for (int t = 0; t < 4; ++t) w0[t] = loadB(0, t);

#pragma unroll 2
  for (int kc = 0; kc < 512; kc += 32) {
    const int kn = (kc + 32) & 511;           // wraps to 0 on last iter (dead)
    v16bf a1 = loadA(kn);
    v16bf w1[4];
#pragma unroll
    for (int t = 0; t < 4; ++t) w1[t] = loadB(kn, t);
#pragma unroll
    for (int t = 0; t < 4; ++t) acc[t] = wmma_bf16(a0, w0[t], acc[t]);
    a0 = a1;
#pragma unroll
    for (int t = 0; t < 4; ++t) w0[t] = w1[t];
  }

  const int halfbase = hf * 8;
#pragma unroll
  for (int t = 0; t < 4; ++t) {
    const int n = colbase + t * 16 + l16;   // column in [0,1536)
    const float bias = qkv_b[n];
    const int which = n >> 9;               // 0=q 1=k 2=v
    const int e = n & 511;
    const int h = e >> 6;
    const int d = e & 63;
#pragma unroll
    for (int j = 0; j < 8; ++j) {
      const int r  = rowbase + halfbase + j;  // global row in [0,8192)
      const int bb = r >> 11;
      const int nn = r & 2047;
      const float v = acc[t][j] + bias;
      const size_t bhi = (size_t)(bb * 8 + h);
      if (which == 0) {
        Qs[(bhi * 2048 + nn) * 64 + d] = (__bf16)(v * 0.125f);  // 1/sqrt(64)
      } else if (which == 1) {
        Ks[(bhi * 2048 + nn) * 64 + d] = (__bf16)v;
      } else {
        VT[(bhi * 64 + d) * 2048 + nn] = (__bf16)v;
      }
    }
  }
}

// ---------------------------------------------------------------------------
// Kernel B: flash attention.  One wave owns 16 query rows of one (b,h);
// block = 4 waves = 64 rows; grid = B*H*(N/64) = 1024 blocks.
// Per 32-key tile: all K+V fragment loads issue first (loads return in
// order, so the S-WMMAs need only a partial loadcnt wait and the V loads
// drain behind WMMA+softmax), then 4 WMMA for S = Q K^T, online softmax,
// P re-layout through per-wave LDS, 4 WMMA for O += P V.
// ---------------------------------------------------------------------------
#define PSTRIDE 40   // floats per LDS row: 16B aligned, spread across banks

__global__ __launch_bounds__(128) void attn_kernel(
    const __bf16* __restrict__ Qs, const __bf16* __restrict__ Ks,
    const __bf16* __restrict__ VT, __bf16* __restrict__ Os) {
  __shared__ float sP[4][16 * PSTRIDE];

  const int lane = threadIdx.x & 31;
  const int wave = threadIdx.x >> 5;
  const int l16  = lane & 15;
  const int hf   = lane >> 4;
  const int b0   = hf * 16, b1 = hf * 16 + 8;
  const int halfbase = hf * 8;

  const int bh      = blockIdx.x >> 5;        // b*8 + h
  const int rowbase = (blockIdx.x & 31) * 64 + wave * 16;  // query row in N
  const size_t base = (size_t)bh * 2048 * 64; // per-(b,h) bf16 element offset

  // Q fragments (2 d-chunks of 32), resident for the whole loop.
  const __bf16* qrow = Qs + base + (size_t)(rowbase + l16) * 64;
  v16bf qf0 = ld_bf16_2x8(qrow + b0, qrow + b1);
  v16bf qf1 = ld_bf16_2x8(qrow + 32 + b0, qrow + 32 + b1);

  float mst[8], lst[8];
#pragma unroll
  for (int j = 0; j < 8; ++j) { mst[j] = -__builtin_inff(); lst[j] = 0.f; }
  v8f acc[4] = {{}, {}, {}, {}};
  float* myP = sP[wave];

  for (int kb = 0; kb < 2048; kb += 32) {
    // Prefetch the next K/V tiles (L2-resident stream -> WGP cache).
    __builtin_prefetch(Ks + base + (size_t)(kb + 32 + l16) * 64, 0, 0);
    __builtin_prefetch(VT + base + (size_t)l16 * 2048 + kb + 32, 0, 0);

    // ---- issue ALL fragment loads for this tile up front ------------------
    const __bf16* k0r = Ks + base + (size_t)(kb + l16) * 64 + hf * 16;
    const __bf16* k1r = Ks + base + (size_t)(kb + 16 + l16) * 64 + hf * 16;
    v16bf kf00 = ld_bf16_16(k0r);
    v16bf kf01 = ld_bf16_16(k0r + 32);
    v16bf kf10 = ld_bf16_16(k1r);
    v16bf kf11 = ld_bf16_16(k1r + 32);
    v16bf vf[4];
#pragma unroll
    for (int t = 0; t < 4; ++t)
      vf[t] = ld_bf16_16(VT + base + (size_t)(t * 16 + l16) * 2048 + kb +
                         hf * 16);

    // ---- S = Q K^T for two 16-key subtiles --------------------------------
    v8f s0 = {}, s1 = {};
    s0 = wmma_bf16(qf0, kf00, s0);
    s0 = wmma_bf16(qf1, kf01, s0);
    s1 = wmma_bf16(qf0, kf10, s1);
    s1 = wmma_bf16(qf1, kf11, s1);

    // ---- online softmax stats (row reductions across 16-lane halves) -----
    float mnew[8], scale[8];
#pragma unroll
    for (int j = 0; j < 8; ++j) {
      float t = fmaxf(s0[j], s1[j]);
#pragma unroll
      for (int m = 1; m < 16; m <<= 1) t = fmaxf(t, __shfl_xor(t, m, 32));
      mnew[j]  = fmaxf(mst[j], t);
      scale[j] = __expf(mst[j] - mnew[j]);
    }
#pragma unroll
    for (int j = 0; j < 8; ++j) {
      const float p0 = __expf(s0[j] - mnew[j]);
      const float p1 = __expf(s1[j] - mnew[j]);
      const int m = halfbase + j;                 // score row
      myP[m * PSTRIDE + l16]      = p0;           // col = key-in-tile
      myP[m * PSTRIDE + 16 + l16] = p1;
      float rs = p0 + p1;
#pragma unroll
      for (int mm = 1; mm < 16; mm <<= 1) rs += __shfl_xor(rs, mm, 32);
      lst[j] = lst[j] * scale[j] + rs;
      mst[j] = mnew[j];
    }
#pragma unroll
    for (int t = 0; t < 4; ++t)
#pragma unroll
      for (int j = 0; j < 8; ++j) acc[t][j] *= scale[j];

    // ---- P (16x32) as bf16 A-fragment via LDS transpose -------------------
    // Same-wave DS ops execute in order, so write->read needs no barrier.
    v16bf pf;
    {
      const float* pr = myP + l16 * PSTRIDE;      // lane = P row
#pragma unroll
      for (int i = 0; i < 8; ++i) pf[i] = (__bf16)pr[b0 + i];
#pragma unroll
      for (int i = 0; i < 8; ++i) pf[8 + i] = (__bf16)pr[b1 + i];
    }
    // ---- O += P @ V  (V fragments already resident) ------------------------
#pragma unroll
    for (int t = 0; t < 4; ++t) acc[t] = wmma_bf16(pf, vf[t], acc[t]);
  }

  // ---- epilogue: normalize and store O as bf16 [B,N,E] --------------------
  const int bb = bh >> 3, h = bh & 7;
#pragma unroll
  for (int t = 0; t < 4; ++t) {
    const int d = t * 16 + l16;
#pragma unroll
    for (int j = 0; j < 8; ++j) {
      const int row = rowbase + halfbase + j;
      const float o = acc[t][j] / lst[j];
      Os[((size_t)(bb * 2048 + row)) * 512 + h * 64 + d] = (__bf16)o;
    }
  }
}

// ---------------------------------------------------------------------------
// Kernel C: out = O @ proj_w^T + proj_b   (M=8192, Nc=512, K=512), f32 out.
// Same double-buffered structure as kernel A.
// ---------------------------------------------------------------------------
__global__ __launch_bounds__(128) void proj_kernel(
    const __bf16* __restrict__ Os, const __bf16* __restrict__ Wp,
    const float* __restrict__ proj_b, float* __restrict__ out) {
  const int lane = threadIdx.x & 31;
  const int wave = threadIdx.x >> 5;
  const int l16  = lane & 15;
  const int hf   = lane >> 4;
  const int b0   = hf * 16, b1 = hf * 16 + 8;
  const int rowbase = blockIdx.x * 16;
  const int colbase = blockIdx.y * 256 + wave * 64;

  v8f acc[4] = {{}, {}, {}, {}};
  const __bf16* orow = Os + (size_t)(rowbase + l16) * 512;

  auto loadA = [&](int kc) {
    return ld_bf16_2x8(orow + kc + b0, orow + kc + b1);
  };
  auto loadB = [&](int kc, int t) {
    return ld_bf16_16(Wp + (size_t)(colbase + t * 16 + l16) * 512 + kc +
                      hf * 16);
  };

  v16bf a0 = loadA(0);
  v16bf w0[4];
#pragma unroll
  for (int t = 0; t < 4; ++t) w0[t] = loadB(0, t);

#pragma unroll 2
  for (int kc = 0; kc < 512; kc += 32) {
    const int kn = (kc + 32) & 511;
    v16bf a1 = loadA(kn);
    v16bf w1[4];
#pragma unroll
    for (int t = 0; t < 4; ++t) w1[t] = loadB(kn, t);
#pragma unroll
    for (int t = 0; t < 4; ++t) acc[t] = wmma_bf16(a0, w0[t], acc[t]);
    a0 = a1;
#pragma unroll
    for (int t = 0; t < 4; ++t) w0[t] = w1[t];
  }

  const int halfbase = hf * 8;
#pragma unroll
  for (int t = 0; t < 4; ++t) {
    const int n = colbase + t * 16 + l16;
    const float bias = proj_b[n];
#pragma unroll
    for (int j = 0; j < 8; ++j) {
      const int r = rowbase + halfbase + j;
      out[(size_t)r * 512 + n] = acc[t][j] + bias;
    }
  }
}

// ---------------------------------------------------------------------------
extern "C" void kernel_launch(void* const* d_in, const int* in_sizes, int n_in,
                              void* d_out, int out_size, void* d_ws,
                              size_t ws_size, hipStream_t stream) {
  const float* x      = (const float*)d_in[0];
  const float* qkv_w  = (const float*)d_in[1];
  const float* qkv_b  = (const float*)d_in[2];
  const float* proj_w = (const float*)d_in[3];
  const float* proj_b = (const float*)d_in[4];

  const size_t SEG  = (size_t)4 * 8 * 2048 * 64;  // 4,194,304 elements
  const int    NX   = 4 * 2048 * 512;             // x elements
  const int    NWQ  = 3 * 512 * 512;              // qkv_w elements
  const int    NWP  = 512 * 512;                  // proj_w elements

  __bf16* Qs = (__bf16*)d_ws;
  __bf16* Ks = Qs + SEG;
  __bf16* VT = Ks + SEG;
  __bf16* Os = VT + SEG;
  __bf16* Xb = Os + SEG;
  __bf16* Wq = Xb + (size_t)NX;
  __bf16* Wp = Wq + (size_t)NWQ;

  cvt_bf16_kernel<<<NX / (256 * 4), 256, 0, stream>>>(x, Xb, NX);
  cvt_bf16_kernel<<<NWQ / (256 * 4), 256, 0, stream>>>(qkv_w, Wq, NWQ);
  cvt_bf16_kernel<<<NWP / (256 * 4), 256, 0, stream>>>(proj_w, Wp, NWP);

  qkv_kernel<<<dim3(512, 6), 128, 0, stream>>>(Xb, Wq, qkv_b, Qs, Ks, VT);
  attn_kernel<<<dim3(1024), 128, 0, stream>>>(Qs, Ks, VT, Os);
  proj_kernel<<<dim3(512, 2), 128, 0, stream>>>(Os, Wp, proj_b,
                                                (float*)d_out);
}
